// GMM_58583353917802
// MI455X (gfx1250) — matrix operand order repaired and checked by reading, not compiled, not run on hardware
//
#include <hip/hip_runtime.h>
#include <hip/hip_bf16.h>
#include <math.h>

typedef __attribute__((ext_vector_type(4)))  _Float16 v4h;
typedef __attribute__((ext_vector_type(8)))  _Float16 v8h;
typedef __attribute__((ext_vector_type(16))) _Float16 v16h;
typedef __attribute__((ext_vector_type(8)))  float    v8f;

#define HDIM    128
#define INDIM   512
#define ROWS    256      // rows per workgroup (8 waves x 32 rows, dual M-tiles)
#define LSTRIDE 136      // padded LDS row stride in halves (272B, 16B aligned)
#define KMIX    10
#define ADIM    3

// LDS layout (byte offsets inside dynamic smem)
#define OFF_HA  0u
#define OFF_HB  (OFF_HA + (unsigned)ROWS * LSTRIDE * 2u)          // 69632
#define OFF_W0  (OFF_HB + (unsigned)ROWS * LSTRIDE * 2u)          // 139264
#define OFF_W1  (OFF_W0 + (unsigned)HDIM * LSTRIDE * 2u)          // 174080
#define OFF_PB  (OFF_W1 + (unsigned)HDIM * LSTRIDE * 2u)          // 208896
#define SMEM_BYTES (OFF_PB + (unsigned)ROWS * 80u * 4u)           // 290816

#define PANEL_HALFS 16384     // 128x128 f16 packed [n][k] panel in d_ws
#define NPANELS     13        // 4 (w_in) + 8 (res) + 1 (head, 80x128)

// ---- fragment loaders (CDNA5 16-bit WMMA layouts, CDNA5 ISA 7.12.2) ----
__device__ __forceinline__ v16h frag_a(const _Float16* base, int m, int kb, int hi) {
  const _Float16* p = base + m * LSTRIDE + kb + (hi << 3);
  v8h lo = *(const v8h*)p;
  v8h hh = *(const v8h*)(p + 16);
  v16h r;
#pragma unroll
  for (int i = 0; i < 8; ++i) { r[i] = lo[i]; r[i + 8] = hh[i]; }
  return r;
}
__device__ __forceinline__ v16h frag_b(const _Float16* base, int n, int kb, int hi) {
  const _Float16* p = base + n * LSTRIDE + kb + (hi << 3);
  v8h lo = *(const v8h*)p;
  v8h hh = *(const v8h*)(p + 16);
  v16h r;
#pragma unroll
  for (int i = 0; i < 8; ++i) { r[i] = lo[i]; r[i + 8] = hh[i]; }
  return r;
}
__device__ __forceinline__ v8f wmma_f16(v16h a, v16h b, v8f c) {
  return __builtin_amdgcn_wmma_f32_16x16x32_f16(false, a, false, b, (short)0, c, false, false);
}

// Issue async global->LDS copy of one pre-transposed f16 weight panel into wbuf
// (padded stride remap via the per-lane LDS address VGPR). Tracked by ASYNCcnt.
__device__ __forceinline__ void issue_panel(const _Float16* ws, int panel,
                                            unsigned lds_base, int tid) {
  const int nchunks = (panel == 12) ? 1280 : 2048;   // 16B chunks (head = 80 rows)
  unsigned long long gbase =
      (unsigned long long)(uintptr_t)(ws + (size_t)panel * PANEL_HALFS);
#pragma unroll
  for (int c0 = 0; c0 < 2048; c0 += 256) {
    if (c0 >= nchunks) break;
    int c = c0 + tid;
    unsigned row  = (unsigned)c >> 4;                       // 16 chunks per 128-half row
    unsigned lds  = lds_base + row * (LSTRIDE * 2u) + ((unsigned)c & 15u) * 16u;
    unsigned goff = (unsigned)c * 16u;
    asm volatile("global_load_async_to_lds_b128 %0, %1, %2"
                 :: "v"(lds), "v"(goff), "s"(gbase) : "memory");
  }
}
__device__ __forceinline__ void wait_async_barrier() {
  asm volatile("s_wait_asynccnt 0x0" ::: "memory");
  __syncthreads();
}

// Dual M-tile GEMM: acc{0,1}[nt] += A(src rows m0 / m0+16) * B(wb), K=128
template <int NT>
__device__ __forceinline__ void mm_dual(v8f* acc0, v8f* acc1, const _Float16* src,
                                        const _Float16* wb, int m0, int lh, int hi) {
#pragma unroll
  for (int ks = 0; ks < 4; ++ks) {
    v16h a0 = frag_a(src, m0, ks * 32, hi);
    v16h a1 = frag_a(src, m0 + 16, ks * 32, hi);
#pragma unroll
    for (int nt = 0; nt < NT; ++nt) {
      v16h bf = frag_b(wb, (nt << 4) + lh, ks * 32, hi);
      acc0[nt] = wmma_f16(a0, bf, acc0[nt]);
      acc1[nt] = wmma_f16(a1, bf, acc1[nt]);
    }
  }
}

// ---------------- prologue: convert+transpose all weights to f16 panels in d_ws ----------------
__global__ void __launch_bounds__(256)
convert_weights_kernel(const float* __restrict__ w_in, const float* __restrict__ res_w1,
                       const float* __restrict__ res_w2, const float* __restrict__ w_head,
                       _Float16* __restrict__ ws) {
  const int total = 12 * PANEL_HALFS + 80 * HDIM;
  for (int t = blockIdx.x * 256 + threadIdx.x; t < total; t += gridDim.x * 256) {
    float v;
    if (t < 4 * PANEL_HALFS) {                      // w_in panels 0..3
      int p = t >> 14, d = t & (PANEL_HALFS - 1);
      int n = d >> 7, kk = d & 127;
      v = w_in[p * PANEL_HALFS + kk * HDIM + n];
    } else if (t < 12 * PANEL_HALFS) {              // res panels 4..11: (w1,w2) x 4 blocks
      int q = (t - 4 * PANEL_HALFS) >> 14;
      int d = t & (PANEL_HALFS - 1);
      int n = d >> 7, kk = d & 127;
      const float* src = (q & 1) ? res_w2 : res_w1;
      v = src[(q >> 1) * PANEL_HALFS + kk * HDIM + n];
    } else {                                        // head panel 12 (80 x 128, zero-padded)
      int d = t - 12 * PANEL_HALFS;
      int n = d >> 7, kk = d & 127;
      v = (n < 70) ? w_head[kk * 70 + n] : 0.f;
    }
    ws[t] = (_Float16)v;
  }
}

// ---------------- main fused kernel ----------------
__global__ void __launch_bounds__(256)
gmm_mlp_kernel(const float* __restrict__ x,      const _Float16* __restrict__ ws,
               const float* __restrict__ b_in,   const float* __restrict__ res_b1,
               const float* __restrict__ res_b2, const float* __restrict__ b_head,
               const float* __restrict__ gumbel, const float* __restrict__ eps,
               const int* __restrict__ det_p,    float* __restrict__ out)
{
  extern __shared__ char smem[];
  _Float16* hA    = (_Float16*)(smem + OFF_HA);   // 256x136 f16 activations (ping)
  _Float16* hB    = (_Float16*)(smem + OFF_HB);   // 256x136 f16 activations (pong)
  _Float16* wbuf0 = (_Float16*)(smem + OFF_W0);   // 128x136 f16 weight panel (even panels)
  _Float16* wbuf1 = (_Float16*)(smem + OFF_W1);   // 128x136 f16 weight panel (odd panels)
  float*    pbuf  = (float*)(smem + OFF_PB);      // 256x80 f32 head params

  const int tid  = threadIdx.x;
  const int wave = tid >> 5;
  const int lane = tid & 31;
  const int lh   = lane & 15;
  const int hi   = lane >> 4;
  const int row0 = blockIdx.x * ROWS;
  const int m0   = (wave << 5) + lh;   // first M-tile row; second is m0+16

  // kick off panel 0 (even -> wbuf0) while we stage x
  issue_panel(ws, 0, OFF_W0, tid);

  // ---------------- layer 1: h = relu(x @ w_in + b_in), K = 512 in 4 panels ----------------
  v8f acc0[8], acc1[8];
#pragma unroll
  for (int nt = 0; nt < 8; ++nt) { v8f z = {}; acc0[nt] = z; acc1[nt] = z; }

#pragma unroll
  for (int p = 0; p < 4; ++p) {   // fully unrolled: all buffer parities are literal
    // stage x panel (f32 -> f16), wave-private 32 rows, vectorized
    const float* xp = x + (size_t)row0 * INDIM + p * 128;
    for (int i = lane; i < 32 * 32; i += 32) {      // (row, float4-chunk)
      int r = i >> 5, q = i & 31;
      float4 v = *(const float4*)(xp + (size_t)((wave << 5) + r) * INDIM + q * 4);
      v4h h4;
      h4[0] = (_Float16)v.x; h4[1] = (_Float16)v.y;
      h4[2] = (_Float16)v.z; h4[3] = (_Float16)v.w;
      *(v4h*)(&hA[((wave << 5) + r) * LSTRIDE + q * 4]) = h4;
    }
    wait_async_barrier();                            // panel p resident
    issue_panel(ws, p + 1, ((p + 1) & 1) ? OFF_W1 : OFF_W0, tid);
    mm_dual<8>(acc0, acc1, hA, (p & 1) ? wbuf1 : wbuf0, m0, lh, hi);
  }
  // epilogue: +bias, relu, write h into hA (wave-private)
#pragma unroll
  for (int nt = 0; nt < 8; ++nt) {
    int col = (nt << 4) + lh;
    float bias = b_in[col];
#pragma unroll
    for (int j = 0; j < 8; ++j) {
      int r0 = (wave << 5) + j + (hi << 3);
      float v0 = acc0[nt][j] + bias; v0 = v0 > 0.f ? v0 : 0.f;
      float v1 = acc1[nt][j] + bias; v1 = v1 > 0.f ? v1 : 0.f;
      hA[r0 * LSTRIDE + col] = (_Float16)v0;
      hA[(r0 + 16) * LSTRIDE + col] = (_Float16)v1;
    }
  }

  // ---------------- 4 residual blocks (panels 4..11; even->wbuf0, odd->wbuf1) ----------------
  for (int blk = 0; blk < 4; ++blk) {
    const int cur = 4 + 2 * blk;                 // even panel
    // t = relu(h @ w1 + b1) : hA -> hB, weights in wbuf0
    wait_async_barrier();
    issue_panel(ws, cur + 1, OFF_W1, tid);       // odd panel -> wbuf1
#pragma unroll
    for (int nt = 0; nt < 8; ++nt) { v8f z = {}; acc0[nt] = z; acc1[nt] = z; }
    mm_dual<8>(acc0, acc1, hA, wbuf0, m0, lh, hi);
    {
      const float* b1 = res_b1 + blk * HDIM;
#pragma unroll
      for (int nt = 0; nt < 8; ++nt) {
        int col = (nt << 4) + lh;
        float bias = b1[col];
#pragma unroll
        for (int j = 0; j < 8; ++j) {
          int r0 = (wave << 5) + j + (hi << 3);
          float v0 = acc0[nt][j] + bias; v0 = v0 > 0.f ? v0 : 0.f;
          float v1 = acc1[nt][j] + bias; v1 = v1 > 0.f ? v1 : 0.f;
          hB[r0 * LSTRIDE + col] = (_Float16)v0;
          hB[(r0 + 16) * LSTRIDE + col] = (_Float16)v1;
        }
      }
    }
    // h = h + t @ w2 + b2 : hB -> hA, weights in wbuf1
    wait_async_barrier();
    issue_panel(ws, cur + 2, OFF_W0, tid);       // next even panel (max 12) -> wbuf0
#pragma unroll
    for (int nt = 0; nt < 8; ++nt) { v8f z = {}; acc0[nt] = z; acc1[nt] = z; }
    mm_dual<8>(acc0, acc1, hB, wbuf1, m0, lh, hi);
    {
      const float* b2 = res_b2 + blk * HDIM;
#pragma unroll
      for (int nt = 0; nt < 8; ++nt) {
        int col = (nt << 4) + lh;
        float bias = b2[col];
#pragma unroll
        for (int j = 0; j < 8; ++j) {
          int r0 = (wave << 5) + j + (hi << 3);
          float v0 = acc0[nt][j] + bias + (float)hA[r0 * LSTRIDE + col];
          float v1 = acc1[nt][j] + bias + (float)hA[(r0 + 16) * LSTRIDE + col];
          hA[r0 * LSTRIDE + col] = (_Float16)v0;
          hA[(r0 + 16) * LSTRIDE + col] = (_Float16)v1;
        }
      }
    }
  }

  // ---------------- head: params = h @ w_head + b_head (panel 12 -> wbuf0, 80 cols) -------
  wait_async_barrier();
#pragma unroll
  for (int nt = 0; nt < 5; ++nt) { v8f z = {}; acc0[nt] = z; acc1[nt] = z; }
  mm_dual<5>(acc0, acc1, hA, wbuf0, m0, lh, hi);
#pragma unroll
  for (int nt = 0; nt < 5; ++nt) {
    int col = (nt << 4) + lh;
    float bias = (col < 70) ? b_head[col] : 0.f;
#pragma unroll
    for (int j = 0; j < 8; ++j) {
      int r0 = (wave << 5) + j + (hi << 3);
      pbuf[r0 * 80 + col] = acc0[nt][j] + bias;
      pbuf[(r0 + 16) * 80 + col] = acc1[nt][j] + bias;
    }
  }
  __syncthreads();   // pbuf rows read across lanes below

  // ---------------- per-row GMM sampling epilogue (1 row per thread) ----------------
  {
    const int g = row0 + tid;
    const float* pr = pbuf + tid * 80;
    const int det = det_p[0];
    const float* gm = gumbel + (size_t)g * KMIX;
    int best = 0;
    float bv = -3.4e38f;
#pragma unroll
    for (int k = 0; k < KMIX; ++k) {
      float lv = pr[60 + k] + (det ? 0.f : gm[k]);   // argmax(logits [+ gumbel])
      if (lv > bv) { bv = lv; best = k; }
    }
#pragma unroll
    for (int a = 0; a < ADIM; ++a) {
      float mn = pr[best * 3 + a];
      float lc = pr[30 + best * 3 + a];
      lc = fminf(fmaxf(lc, -20.f), 2.f);
      float act = det ? mn : (mn + sqrtf(expf(lc)) * eps[(size_t)g * 3 + a]);
      float s = (a == 0) ? 6.f : ((a == 1) ? 6.f : 3.14159265358979323846f);
      out[(size_t)g * 3 + a] = s * tanhf(act);
    }
  }
}

extern "C" void kernel_launch(void* const* d_in, const int* in_sizes, int n_in,
                              void* d_out, int out_size, void* d_ws, size_t ws_size,
                              hipStream_t stream) {
  const float* x      = (const float*)d_in[0];
  const float* w_in   = (const float*)d_in[1];
  const float* b_in   = (const float*)d_in[2];
  const float* res_w1 = (const float*)d_in[3];
  const float* res_b1 = (const float*)d_in[4];
  const float* res_w2 = (const float*)d_in[5];
  const float* res_b2 = (const float*)d_in[6];
  const float* w_head = (const float*)d_in[7];
  const float* b_head = (const float*)d_in[8];
  const float* gumbel = (const float*)d_in[9];
  const float* eps    = (const float*)d_in[10];
  const int*   det    = (const int*)d_in[11];
  float*       out    = (float*)d_out;
  _Float16*    ws     = (_Float16*)d_ws;   // 13 panels, ~404 KB

  const int B = in_sizes[0] / INDIM;  // 262144

  convert_weights_kernel<<<dim3(808), dim3(256), 0, stream>>>(
      w_in, res_w1, res_w2, w_head, ws);

  hipFuncSetAttribute((const void*)gmm_mlp_kernel,
                      hipFuncAttributeMaxDynamicSharedMemorySize, (int)SMEM_BYTES);
  gmm_mlp_kernel<<<dim3(B / ROWS), dim3(256), SMEM_BYTES, stream>>>(
      x, ws, b_in, res_b1, res_b2, b_head, gumbel, eps, det, out);
}
